// InterMA_49100066128305
// MI455X (gfx1250) — matrix operand order repaired and checked by reading, not compiled, not run on hardware
//
#include <hip/hip_runtime.h>
#include <hip/hip_bf16.h>

#define Nn 4096
#define Mm 2048
#define Dd 256
#define Ee 64
#define Hh 8
#define HDd 512

typedef _Float16 half_t;
typedef __attribute__((ext_vector_type(16))) _Float16 v16h;
typedef __attribute__((ext_vector_type(8)))  _Float16 v8h;
typedef __attribute__((ext_vector_type(8)))  float    v8f;

union AFrag {
    v16h v;
    v8h  h8[2];
    half_t e[16];
};

#define INF __builtin_inff()

// ---------------- WMMA helper (CDNA5 v_wmma_f32_16x16x32_f16) ----------------

__device__ inline v8f wmma16(v16h a, v16h b, v8f c) {
    return __builtin_amdgcn_wmma_f32_16x16x32_f16(false, a, false, b, (short)0, c, false, false);
}

// A fragment: A row-major [rows x lda], tile rows row0..+15, K block k0..+31.
// Element j of lane l: K = k0 + (j<8 ? j : j+8) + 8*(l>=16), row = row0 + (l&15).
// -> two contiguous 16-byte loads per lane.
__device__ inline v16h load_a16(const half_t* __restrict__ A, int lda, int row0, int k0, int lane) {
    int r  = row0 + (lane & 15);
    int kh = (lane >> 4) & 1;
    const half_t* p = A + (size_t)r * lda + k0 + 8 * kh;
    AFrag f;
    f.h8[0] = *(const v8h*)p;
    f.h8[1] = *(const v8h*)(p + 16);
    return f.v;
}

// B fragment from K-contiguous (transposed) storage: Bt row-major [cols x ldbt],
// logical B[k][c] = Bt[c][k]. Element j of lane l: K = k0 + (j<8 ? j : j+8) + 8*(l>=16),
// col = col0 + (l&15) -> two contiguous 16-byte loads per lane.
__device__ inline v16h load_bT16(const half_t* __restrict__ Bt, int ldbt, int k0, int col0, int lane) {
    int c  = col0 + (lane & 15);
    int kh = (lane >> 4) & 1;
    const half_t* p = Bt + (size_t)c * ldbt + k0 + 8 * kh;
    AFrag f;
    f.h8[0] = *(const v8h*)p;
    f.h8[1] = *(const v8h*)(p + 16);
    return f.v;
}

// ---------------- LayerNorm -> f16 ----------------

__global__ void ln_f16_kernel(const float* __restrict__ x, const float* __restrict__ g,
                              const float* __restrict__ b, half_t* __restrict__ y) {
    __shared__ float red[256];
    int row = blockIdx.x;
    int t   = threadIdx.x;
    float v = x[(size_t)row * Dd + t];
    red[t] = v;
    __syncthreads();
    for (int s = 128; s > 0; s >>= 1) {
        if (t < s) red[t] += red[t + s];
        __syncthreads();
    }
    float mu = red[0] * (1.0f / Dd);
    __syncthreads();
    float d = v - mu;
    red[t] = d * d;
    __syncthreads();
    for (int s = 128; s > 0; s >>= 1) {
        if (t < s) red[t] += red[t + s];
        __syncthreads();
    }
    float var = red[0] * (1.0f / Dd);
    float r   = rsqrtf(var + 1e-5f);
    y[(size_t)row * Dd + t] = (half_t)(d * r * g[t] + b[t]);
}

// ---------------- tiled transpose + f32->f16 convert: out[c][r] = (f16) in[r][c] ----------------
// R, C multiples of 32. blockDim = (32, 8). batch via blockIdx.z.

__global__ void transpose_cvt_kernel(const float* __restrict__ in, half_t* __restrict__ out,
                                     int R, int C) {
    __shared__ float tile[32][33];
    int bx = blockIdx.x * 32;   // col block
    int by = blockIdx.y * 32;   // row block
    const float* inb  = in  + (size_t)blockIdx.z * R * C;
    half_t*      outb = out + (size_t)blockIdx.z * R * C;
    int tx = threadIdx.x, ty = threadIdx.y;
#pragma unroll
    for (int i = 0; i < 32; i += 8)
        tile[ty + i][tx] = inb[(size_t)(by + ty + i) * C + bx + tx];
    __syncthreads();
#pragma unroll
    for (int i = 0; i < 32; i += 8)
        outb[(size_t)(bx + ty + i) * R + by + tx] = (half_t)tile[tx][ty + i];
}

// ---------------- Type-dispatched projection ----------------
// A [rows x 256] f16, Wt [T x 512 x 256] f16 (transposed weights), out f16.
// Each wave: 32 rows x 16 cols (two 16x16 subtiles sharing B fragments).
// TOUT=false: out[r][c] (ld = HDd). TOUT=true: out[c][r] (ld = rows), vectorized store.

template <int T, bool TOUT>
__global__ void proj_kernel(const half_t* __restrict__ A, const half_t* __restrict__ Wt,
                            const int* __restrict__ types, half_t* __restrict__ out, int ldo) {
    int lane = threadIdx.x & 31;
    int wave = threadIdx.x >> 5;
    int col0 = (blockIdx.x * 4 + wave) * 16;   // over HD
    int row0 = blockIdx.y * 32;

    v8f acc[2][T];
#pragma unroll
    for (int s = 0; s < 2; ++s)
#pragma unroll
        for (int t = 0; t < T; ++t) acc[s][t] = (v8f){};

    for (int k0 = 0; k0 < Dd; k0 += 32) {
        v16h a0 = load_a16(A, Dd, row0, k0, lane);
        v16h a1 = load_a16(A, Dd, row0 + 16, k0, lane);
#pragma unroll
        for (int t = 0; t < T; ++t) {
            v16h b = load_bT16(Wt + (size_t)t * HDd * Dd, Dd, k0, col0, lane);
            acc[0][t] = wmma16(a0, b, acc[0][t]);
            acc[1][t] = wmma16(a1, b, acc[1][t]);
        }
    }

    int c  = col0 + (lane & 15);
    int kh = (lane >> 4) & 1;
#pragma unroll
    for (int s = 0; s < 2; ++s) {
        int rbase = row0 + s * 16 + kh * 8;
        if (TOUT) {
            v8h tmp;
#pragma unroll
            for (int i = 0; i < 8; ++i) {
                int r  = rbase + i;
                int ty = types[r];
                float val = acc[s][0][i];
#pragma unroll
                for (int t = 1; t < T; ++t)
                    if (ty == t) val = acc[s][t][i];
                tmp[i] = (half_t)val;
            }
            *(v8h*)&out[(size_t)c * ldo + rbase] = tmp;
        } else {
#pragma unroll
            for (int i = 0; i < 8; ++i) {
                int r  = rbase + i;
                int ty = types[r];
                float val = acc[s][0][i];
#pragma unroll
                for (int t = 1; t < T; ++t)
                    if (ty == t) val = acc[s][t][i];
                out[(size_t)r * HDd + c] = (half_t)val;
            }
        }
    }
}

// ---------------- scores[h,n,m] = mask(q_h . k_h / 8) -> d_out ----------------
// Each wave: 32 n x 16 m, ALL 8 heads. The incidence tile is loaded once,
// compared, and kept as a packed 16-bit per-lane mask reused across heads
// (8x reduction of incidence HBM traffic: 256 MB -> 32 MB).

__global__ void scores_kernel(const half_t* __restrict__ q, const half_t* __restrict__ k,
                              const int* __restrict__ inc, float* __restrict__ sc) {
    int lane = threadIdx.x & 31;
    int wave = threadIdx.x >> 5;
    int m0 = (blockIdx.x * 4 + wave) * 16;
    int n0 = blockIdx.y * 32;

    int m   = m0 + (lane & 15);
    int khh = (lane >> 4) & 1;

    // per-lane packed incidence mask: bits 0..7 -> subtile0 rows, 8..15 -> subtile1
    unsigned mask = 0;
#pragma unroll
    for (int i = 0; i < 8; ++i) {
        mask |= (inc[(size_t)(n0 + khh * 8 + i) * Mm + m]      != 0) ? (1u << i)       : 0u;
        mask |= (inc[(size_t)(n0 + 16 + khh * 8 + i) * Mm + m] != 0) ? (1u << (8 + i)) : 0u;
    }

    const float scale = 0.125f;   // 1/sqrt(64)
    for (int h = 0; h < Hh; ++h) {
        const half_t* qh = q + h * Ee;   // [N x 512] head slice
        const half_t* kh = k + h * Ee;   // [M x 512] head slice (K-contiguous per m)

        v8f acc0 = (v8f){}, acc1 = (v8f){};
#pragma unroll
        for (int k0 = 0; k0 < Ee; k0 += 32) {
            v16h b  = load_bT16(kh, HDd, k0, m0, lane);
            v16h a0 = load_a16(qh, HDd, n0, k0, lane);
            v16h a1 = load_a16(qh, HDd, n0 + 16, k0, lane);
            acc0 = wmma16(a0, b, acc0);
            acc1 = wmma16(a1, b, acc1);
        }

#pragma unroll
        for (int i = 0; i < 8; ++i) {
            int n = n0 + khh * 8 + i;
            float s = ((mask >> i) & 1u) ? acc0[i] * scale : -INF;
            sc[((size_t)h * Nn + n) * Mm + m] = s;
        }
#pragma unroll
        for (int i = 0; i < 8; ++i) {
            int n = n0 + 16 + khh * 8 + i;
            float s = ((mask >> (8 + i)) & 1u) ? acc1[i] * scale : -INF;
            sc[((size_t)h * Nn + n) * Mm + m] = s;
        }
    }
}

// ---------------- softmax row stats (max, sumexp) over M ----------------

__global__ void softmax_stats_kernel(const float* __restrict__ sc, float* __restrict__ rmax,
                                     float* __restrict__ rsum) {
    __shared__ float red[256];
    size_t row = blockIdx.x;   // h*N + n
    const float* p = sc + row * Mm;
    int t = threadIdx.x;

    float mx = -INF;
    for (int j = t; j < Mm; j += 256) mx = fmaxf(mx, p[j]);
    red[t] = mx;
    __syncthreads();
    for (int s = 128; s > 0; s >>= 1) {
        if (t < s) red[t] = fmaxf(red[t], red[t + s]);
        __syncthreads();
    }
    mx = red[0];
    __syncthreads();

    float sum = 0.f;
    if (mx > -INF) {
        for (int j = t; j < Mm; j += 256) {
            float v = p[j];
            if (v > -INF) sum += __expf(v - mx);
        }
    }
    red[t] = sum;
    __syncthreads();
    for (int s = 128; s > 0; s >>= 1) {
        if (t < s) red[t] += red[t + s];
        __syncthreads();
    }
    if (t == 0) {
        rmax[row] = mx;
        rsum[row] = red[0];
    }
}

// ---------------- att[n, h*64+e] = silu( softmax(scores) @ v ) ----------------
// Block = 4 waves covering E=64 for one (h, 16-row n tile). The softmax-weight
// tile is computed ONCE per block into LDS (coalesced f32 score reads, exp once),
// then each wave builds A fragments with two ds_load_b128. B = v^T, K-contiguous.

#define SEG   1024
#define WPAD  (SEG + 8)   // pad 16B per row -> conflict-free ds_load_b128

__global__ void av_kernel(const float* __restrict__ sc, const half_t* __restrict__ vT,
                          const float* __restrict__ rmax, const float* __restrict__ rsum,
                          half_t* __restrict__ att) {
    __shared__ half_t wtile[16 * WPAD];   // 33 KB
    __shared__ float  mxs[16], invs[16];

    int tid  = threadIdx.x;
    int lane = tid & 31;
    int wave = tid >> 5;         // 0..3 -> e tile
    int h    = blockIdx.y;
    int n0   = blockIdx.x * 16;
    int e0   = wave * 16;

    if (tid < 16) {
        float mx = rmax[(size_t)h * Nn + n0 + tid];
        float s  = rsum[(size_t)h * Nn + n0 + tid];
        mxs[tid]  = mx;
        invs[tid] = (s > 0.f) ? 1.f / s : 0.f;
    }
    __syncthreads();

    int kh = (lane >> 4) & 1;
    int ar = lane & 15;
    v8f acc = (v8f){};

    for (int seg = 0; seg < Mm; seg += SEG) {
        // cooperatively build 16 x SEG f16 weight tile
        for (int idx = tid; idx < 16 * SEG; idx += 128) {
            int r  = idx >> 10;          // SEG = 1024
            int kk = idx & (SEG - 1);
            float s = sc[((size_t)h * Nn + n0 + r) * Mm + seg + kk];
            float mx = mxs[r], inv = invs[r];
            float w = (s > -INF && inv != 0.f) ? __expf(s - mx) * inv : 0.f;
            wtile[r * WPAD + kk] = (half_t)w;
        }
        __syncthreads();

        for (int k0 = 0; k0 < SEG; k0 += 32) {
            AFrag fa;
            const half_t* p = &wtile[ar * WPAD + k0 + 8 * kh];
            fa.h8[0] = *(const v8h*)p;
            fa.h8[1] = *(const v8h*)(p + 16);
            v16h b = load_bT16(vT + (size_t)h * Ee * Mm, Mm, seg + k0, e0, lane);
            acc = wmma16(fa.v, b, acc);
        }
        __syncthreads();
    }

    int e  = e0 + (lane & 15);
    int nb = n0 + kh * 8;
#pragma unroll
    for (int i = 0; i < 8; ++i) {
        int n   = nb + i;
        float x = acc[i];
        float y = x / (1.f + __expf(-x));   // SiLU
        att[(size_t)n * HDd + h * Ee + e] = (half_t)y;
    }
}

// ---------------- final = concat(node, att) @ W_out + b_out ----------------
// WoutT: [256 x 768] f16 (transposed). Two 16-row subtiles per wave.

__global__ void final_kernel(const half_t* __restrict__ node16, const half_t* __restrict__ att,
                             const half_t* __restrict__ WoutT, const float* __restrict__ bout,
                             float* __restrict__ out) {
    int lane = threadIdx.x & 31;
    int wave = threadIdx.x >> 5;
    int col0 = (blockIdx.x * 4 + wave) * 16;   // over D
    int row0 = blockIdx.y * 32;

    const int KT = HDd + Dd;   // 768
    v8f acc0 = (v8f){}, acc1 = (v8f){};
    for (int k0 = 0; k0 < Dd; k0 += 32) {
        v16h b  = load_bT16(WoutT, KT, k0, col0, lane);
        v16h a0 = load_a16(node16, Dd, row0, k0, lane);
        v16h a1 = load_a16(node16, Dd, row0 + 16, k0, lane);
        acc0 = wmma16(a0, b, acc0);
        acc1 = wmma16(a1, b, acc1);
    }
    for (int k0 = 0; k0 < HDd; k0 += 32) {
        v16h b  = load_bT16(WoutT, KT, Dd + k0, col0, lane);
        v16h a0 = load_a16(att, HDd, row0, k0, lane);
        v16h a1 = load_a16(att, HDd, row0 + 16, k0, lane);
        acc0 = wmma16(a0, b, acc0);
        acc1 = wmma16(a1, b, acc1);
    }

    int c  = col0 + (lane & 15);
    int kh = (lane >> 4) & 1;
    float bias = bout[c];
#pragma unroll
    for (int i = 0; i < 8; ++i)
        out[(size_t)(row0 + kh * 8 + i) * Dd + c] = acc0[i] + bias;
#pragma unroll
    for (int i = 0; i < 8; ++i)
        out[(size_t)(row0 + 16 + kh * 8 + i) * Dd + c] = acc1[i] + bias;
}

// ---------------- host-side launch ----------------

static inline size_t align256(size_t x) { return (x + 255) & ~(size_t)255; }

extern "C" void kernel_launch(void* const* d_in, const int* in_sizes, int n_in,
                              void* d_out, int out_size, void* d_ws, size_t ws_size,
                              hipStream_t stream) {
    (void)in_sizes; (void)n_in; (void)out_size; (void)ws_size;

    const float* edge_in = (const float*)d_in[0];   // [M, D]
    const float* node_in = (const float*)d_in[1];   // [N, D]
    const int*   inc     = (const int*)  d_in[2];   // [N, M]
    const int*   ntype   = (const int*)  d_in[3];   // [N]
    const int*   etype   = (const int*)  d_in[4];   // [M]
    const float* Wq      = (const float*)d_in[5];   // [2, D, HD]
    const float* Wk      = (const float*)d_in[6];   // [4, D, HD]
    const float* Wv      = (const float*)d_in[7];   // [4, D, HD]
    const float* Wout    = (const float*)d_in[8];   // [HD+D, D]
    const float* bout    = (const float*)d_in[9];   // [D]
    const float* ln_e_g  = (const float*)d_in[10];
    const float* ln_e_b  = (const float*)d_in[11];
    const float* ln_n_g  = (const float*)d_in[12];
    const float* ln_n_b  = (const float*)d_in[13];

    float* outF   = (float*)d_out;                  // [N, D]
    float* scores = outF + (size_t)Nn * Dd;         // [H, N, M]

    // workspace carve-up (~19 MB)
    char* w = (char*)d_ws;
    size_t off = 0;
    half_t* node16 = (half_t*)(w + off); off = align256(off + (size_t)Nn * Dd * 2);
    half_t* edge16 = (half_t*)(w + off); off = align256(off + (size_t)Mm * Dd * 2);
    half_t* q16    = (half_t*)(w + off); off = align256(off + (size_t)Nn * HDd * 2);
    half_t* k16    = (half_t*)(w + off); off = align256(off + (size_t)Mm * HDd * 2);
    half_t* vT16   = (half_t*)(w + off); off = align256(off + (size_t)HDd * Mm * 2);
    half_t* att16  = (half_t*)(w + off); off = align256(off + (size_t)Nn * HDd * 2);
    half_t* WqT    = (half_t*)(w + off); off = align256(off + (size_t)2 * Dd * HDd * 2);
    half_t* WkT    = (half_t*)(w + off); off = align256(off + (size_t)4 * Dd * HDd * 2);
    half_t* WvT    = (half_t*)(w + off); off = align256(off + (size_t)4 * Dd * HDd * 2);
    half_t* WoutT  = (half_t*)(w + off); off = align256(off + (size_t)(HDd + Dd) * Dd * 2);
    float*  rmax   = (float*)(w + off);  off = align256(off + (size_t)Hh * Nn * 4);
    float*  rsum   = (float*)(w + off);  off = align256(off + (size_t)Hh * Nn * 4);

    // 1. LayerNorms -> f16
    ln_f16_kernel<<<Mm, 256, 0, stream>>>(edge_in, ln_e_g, ln_e_b, edge16);
    ln_f16_kernel<<<Nn, 256, 0, stream>>>(node_in, ln_n_g, ln_n_b, node16);

    // 2. weight transpose + f32 -> f16 (K-contiguous B operands)
    transpose_cvt_kernel<<<dim3(HDd / 32, Dd / 32, 2), dim3(32, 8), 0, stream>>>(Wq, WqT, Dd, HDd);
    transpose_cvt_kernel<<<dim3(HDd / 32, Dd / 32, 4), dim3(32, 8), 0, stream>>>(Wk, WkT, Dd, HDd);
    transpose_cvt_kernel<<<dim3(HDd / 32, Dd / 32, 4), dim3(32, 8), 0, stream>>>(Wv, WvT, Dd, HDd);
    transpose_cvt_kernel<<<dim3(Dd / 32, (HDd + Dd) / 32, 1), dim3(32, 8), 0, stream>>>(Wout, WoutT, HDd + Dd, Dd);

    // 3. type-dispatched projections (WMMA); v written transposed for the AV pass
    proj_kernel<2, false><<<dim3(HDd / 64, Nn / 32), 128, 0, stream>>>(node16, WqT, ntype, q16, HDd);
    proj_kernel<4, false><<<dim3(HDd / 64, Mm / 32), 128, 0, stream>>>(edge16, WkT, etype, k16, HDd);
    proj_kernel<4, true ><<<dim3(HDd / 64, Mm / 32), 128, 0, stream>>>(edge16, WvT, etype, vT16, Mm);

    // 4. masked scores for all heads -> d_out (WMMA, incidence loaded once per tile)
    scores_kernel<<<dim3(Mm / 64, Nn / 32), 128, 0, stream>>>(q16, k16, inc, scores);

    // 5. softmax row stats
    softmax_stats_kernel<<<Hh * Nn, 256, 0, stream>>>(scores, rmax, rsum);

    // 6. attention output + SiLU (WMMA, LDS-shared softmax weights)
    av_kernel<<<dim3(Nn / 16, Hh), 128, 0, stream>>>(scores, vT16, rmax, rsum, att16);

    // 7. final projection + bias (WMMA)
    final_kernel<<<dim3(Dd / 64, Nn / 32), 128, 0, stream>>>(node16, att16, WoutT, bout, outF);
}
